// dgl_HNN_43379169689826
// MI455X (gfx1250) — matrix operand order repaired and checked by reading, not compiled, not run on hardware
//
#include <hip/hip_runtime.h>
#include <math.h>

typedef __attribute__((ext_vector_type(16))) _Float16 v16h;
typedef __attribute__((ext_vector_type(8)))  float    v8f;

#define FEATS 64

// ---------------------------------------------------------------- utilities
__global__ void zero_f4_kernel(float* __restrict__ p, int n4) {
    int i = blockIdx.x * blockDim.x + threadIdx.x;
    if (i < n4) ((float4*)p)[i] = make_float4(0.f, 0.f, 0.f, 0.f);
}

// count degrees with f32 atomics (values are small integers -> exact in f32)
__global__ void degree_kernel(const int* __restrict__ src, const int* __restrict__ dst,
                              float* __restrict__ deg_out, float* __restrict__ deg_in, int n_edges) {
    int e = blockIdx.x * blockDim.x + threadIdx.x;
    if (e >= n_edges) return;
    atomicAdd(&deg_out[src[e]], 1.0f);
    atomicAdd(&deg_in[dst[e]], 1.0f);
}

// deg -> deg>0 ? rsqrt(deg) : 0   (applied to both arrays at once: n = 2*N)
__global__ void norm_kernel(float* __restrict__ deg, int n) {
    int i = blockIdx.x * blockDim.x + threadIdx.x;
    if (i >= n) return;
    float d = deg[i];
    deg[i] = (d > 0.f) ? rsqrtf(d) : 0.f;
}

// h[node][f] = x[node][f] * norm_out[node], float4 vectorized (16 float4 per node)
__global__ void scale_kernel(const float* __restrict__ x, const float* __restrict__ norm_out,
                             float* __restrict__ h, int n4) {
    int i = blockIdx.x * blockDim.x + threadIdx.x;
    if (i >= n4) return;
    int node = i >> 4;
    float s = norm_out[node];
    float4 v = ((const float4*)x)[i];
    v.x *= s; v.y *= s; v.z *= s; v.w *= s;
    ((float4*)h)[i] = v;
}

// agg[dst[e]][:] += h[src[e]][:]   — 16 threads per edge, float4 gather + 4 f32 atomics
__global__ void scatter_kernel(const float* __restrict__ h, const int* __restrict__ src,
                               const int* __restrict__ dst, float* __restrict__ agg, int n_edges) {
    int idx = blockIdx.x * blockDim.x + threadIdx.x;
    int e = idx >> 4;
    if (e >= n_edges) return;
    int g = (idx & 15) << 2;           // feature group of 4
    int s = src[e];
    int d = dst[e];
    float4 hv = *(const float4*)(h + s * FEATS + g);
    float* ap = agg + d * FEATS + g;
    atomicAdd(ap + 0, hv.x);
    atomicAdd(ap + 1, hv.y);
    atomicAdd(ap + 2, hv.z);
    atomicAdd(ap + 3, hv.w);
}

// ---------------------------------------------------------------- WMMA GEMM
// Z = (agg @ W) row-scaled by norm_in + bias, for a 16-node block per wave.
// (norm_in is a per-row scalar, so it commutes past the GEMM: applied in f32 epilogue.)
// MODE 0: out = tanh(Z) * norm_out          (layer-1 epilogue, produces h2)
// MODE 1: out = Z @ J.T  (column c -> c^32, sign from bit 5)
//
// W (64x64 row-major [K][N], f32) is staged into LDS pre-swizzled into this kernel's
// contiguous-k B-fragment layout: fragment f = kc*4 + t holds, for each lane,
// 16 contiguous f16 = W[kc*32 + (lane>>4)*16 + i][t*16 + (lane&15)], i=0..15.
// A B operand is then a single aligned 32-byte v16h LDS load (2x ds_load_b128).
template <int MODE>
__global__ void __launch_bounds__(256)
gemm_wmma_kernel(const float* __restrict__ agg, const float* __restrict__ norm_in,
                 const float* __restrict__ W, const float* __restrict__ bias,
                 const float* __restrict__ norm_out, float* __restrict__ out,
                 int n_nodes) {
    __shared__ __align__(32) _Float16 Wl[8 * 512];   // 8 fragments x (32 lanes x 16 halves) = 8 KB

    // Coalesced staging: each thread reads 4 float4 of W, 1024 floats apart (global_load_b128),
    // scatters each element to its swizzled fragment slot (ds_store_b16).
    // widx = K*64 + N  ->  frag = (K>>5)*4 + (N>>4); lane = ((K>>4)&1)*16 + (N&15); i = K&15.
    for (int base = threadIdx.x * 4; base < FEATS * FEATS; base += 256 * 4) {
        float4 w4 = *(const float4*)(W + base);
        const float wv[4] = {w4.x, w4.y, w4.z, w4.w};
        #pragma unroll
        for (int j = 0; j < 4; ++j) {
            int widx = base + j;
            int K = widx >> 6, Ncol = widx & 63;
            int frag = ((K >> 5) << 2) + (Ncol >> 4);
            int ln   = (((K >> 4) & 1) << 4) + (Ncol & 15);
            Wl[frag * 512 + ln * 16 + (K & 15)] = (_Float16)wv[j];
        }
    }
    __syncthreads();

    const int wave = threadIdx.x >> 5;       // 0..7
    const int lane = threadIdx.x & 31;
    const int rowblk = blockIdx.x * 8 + wave;
    if (rowblk * 16 >= n_nodes) return;      // wave-uniform exit: EXEC stays all-ones

    const int m = lane & 15;                 // A row within tile
    const int khalf = lane >> 4;             // which 16-wide K half this lane holds
    const int node0 = rowblk * 16;
    const int nodeA = (node0 + m < n_nodes) ? (node0 + m) : (n_nodes - 1);
    const float* arow_base = agg + nodeA * FEATS;
    const _Float16* wfrag = &Wl[lane * 16];

    v8f acc0 = {}, acc1 = {}, acc2 = {}, acc3 = {};

    #pragma unroll
    for (int kc = 0; kc < 2; ++kc) {         // K chunks of 32
        // A tile 16x32 f16: lanes 0-15 hold K = 0..15; lanes 16-31 hold K = 16..31.
        v16h a;
        const float* arow = arow_base + kc * 32 + khalf * 16;
        #pragma unroll
        for (int i = 0; i < 8; ++i) {
            a[i]     = (_Float16)arow[i];
            a[i + 8] = (_Float16)arow[8 + i];
        }
        acc0 = __builtin_amdgcn_wmma_f32_16x16x32_f16(false, a, false,
                 *(const v16h*)(wfrag + (kc * 4 + 0) * 512), (short)0, acc0, false, false);
        acc1 = __builtin_amdgcn_wmma_f32_16x16x32_f16(false, a, false,
                 *(const v16h*)(wfrag + (kc * 4 + 1) * 512), (short)0, acc1, false, false);
        acc2 = __builtin_amdgcn_wmma_f32_16x16x32_f16(false, a, false,
                 *(const v16h*)(wfrag + (kc * 4 + 2) * 512), (short)0, acc2, false, false);
        acc3 = __builtin_amdgcn_wmma_f32_16x16x32_f16(false, a, false,
                 *(const v16h*)(wfrag + (kc * 4 + 3) * 512), (short)0, acc3, false, false);
    }

    // Epilogue. C/D layout: VGPR r -> M = r + 8*khalf, N = m (per 16-col tile).
    float nin_r[8], nout_r[8];
    #pragma unroll
    for (int r = 0; r < 8; ++r) {
        int nodeM = node0 + r + 8 * khalf;
        int nc = (nodeM < n_nodes) ? nodeM : (n_nodes - 1);
        nin_r[r]  = norm_in[nc];
        nout_r[r] = norm_out[nc];   // unconditional: avoids branchy conditional loads
    }

    const v8f accs[4] = {acc0, acc1, acc2, acc3};
    const bool full = (node0 + 16 <= n_nodes);   // always true when N % 16 == 0
    #pragma unroll
    for (int t = 0; t < 4; ++t) {
        const int col = t * 16 + m;
        const float bv = bias[col];
        const int dcol = (MODE == 0) ? col : (col ^ 32);        // J.T column permute
        const float sgn = (MODE == 0) ? 1.f : ((col & 32) ? 1.f : -1.f);
        #pragma unroll
        for (int r = 0; r < 8; ++r) {
            const int nodeM = node0 + r + 8 * khalf;
            float v = accs[t][r] * nin_r[r] + bv;
            if (MODE == 0) v = tanhf(v) * nout_r[r];   // tanh + next layer's src norm
            else           v = sgn * v;
            if (full || nodeM < n_nodes)
                out[nodeM * FEATS + dcol] = v;
        }
    }
}

// ---------------------------------------------------------------- launcher
extern "C" void kernel_launch(void* const* d_in, const int* in_sizes, int n_in,
                              void* d_out, int out_size, void* d_ws, size_t ws_size,
                              hipStream_t stream) {
    const float* x   = (const float*)d_in[0];
    const int*   src = (const int*)  d_in[1];
    const int*   dst = (const int*)  d_in[2];
    const float* W1  = (const float*)d_in[3];
    const float* b1  = (const float*)d_in[4];
    const float* W2  = (const float*)d_in[5];
    const float* b2  = (const float*)d_in[6];
    float* out = (float*)d_out;

    const int N = in_sizes[0] / FEATS;   // 50000
    const int E = in_sizes[1];           // 800000

    // workspace layout (floats): deg_out[N] | deg_in[N] | h[N*64] | agg[N*64]
    float* ws      = (float*)d_ws;
    float* deg_out = ws;
    float* deg_in  = ws + N;
    float* h       = ws + 2 * N;
    float* agg     = ws + 2 * N + N * FEATS;

    const int TB = 256;
    const int nf4  = N * (FEATS / 4);            // float4 count of a node-feature buffer
    const int rowblks = (N + 15) / 16;
    const int gemm_grid = (rowblks + 7) / 8;

    // 1) degrees -> norms
    zero_f4_kernel<<<((2 * N / 4) + TB - 1) / TB, TB, 0, stream>>>(deg_out, 2 * N / 4);
    degree_kernel<<<(E + TB - 1) / TB, TB, 0, stream>>>(src, dst, deg_out, deg_in, E);
    norm_kernel<<<(2 * N + TB - 1) / TB, TB, 0, stream>>>(deg_out, 2 * N);
    // deg_out/deg_in now hold norm_out/norm_in

    // 2) layer 1: h = x*norm_out ; agg = scatter(h) ; h2 = tanh((agg@W1)*nin+b1)*norm_out
    scale_kernel<<<(nf4 + TB - 1) / TB, TB, 0, stream>>>(x, deg_out, h, nf4);
    zero_f4_kernel<<<(nf4 + TB - 1) / TB, TB, 0, stream>>>(agg, nf4);
    scatter_kernel<<<((E * 16) + TB - 1) / TB, TB, 0, stream>>>(h, src, dst, agg, E);
    gemm_wmma_kernel<0><<<gemm_grid, TB, 0, stream>>>(agg, deg_in, W1, b1, deg_out, h, N);

    // 3) layer 2: agg = scatter(h2) ; out = ((agg@W2)*nin + b2) @ J.T
    zero_f4_kernel<<<(nf4 + TB - 1) / TB, TB, 0, stream>>>(agg, nf4);
    scatter_kernel<<<((E * 16) + TB - 1) / TB, TB, 0, stream>>>(h, src, dst, agg, E);
    gemm_wmma_kernel<1><<<gemm_grid, TB, 0, stream>>>(agg, deg_in, W2, b2, deg_out, out, N);
}
